// JointMamba_79130477461818
// MI455X (gfx1250) — matrix operand order repaired and checked by reading, not compiled, not run on hardware
//
#include <hip/hip_runtime.h>
#include <hip/hip_bf16.h>

// ---------------------------------------------------------------------------
// JointMamba (JEGO 4-direction scan + Mamba mixer) for MI455X / gfx1250
//
// Shapes: B=4, C=256, H=W=96, STEP=2  -> N=16 seqs, L=4608, d_in=512,
// dt_rank=16, D_STATE=16, D_CONV=4.
//
// Precision strategy: all GEMM operands stored f16 (v_wmma_f32_16x16x32_f16,
// f32 accumulate), scan math in f32 registers.
// ---------------------------------------------------------------------------

typedef __attribute__((ext_vector_type(16))) _Float16 v16h;
typedef __attribute__((ext_vector_type(8)))  _Float16 v8h;
typedef __attribute__((ext_vector_type(8)))  float    v8f;

#define BB      4
#define CC      256
#define HH      96
#define WW      96
#define KK      4
#define LL      4608            // (96/2)*(192/2)
#define NN      16              // B*K
#define DIN     512
#define DTR     16
#define DST     16
#define NTOK    (NN * LL)       // 73728 tokens

// ---------------------------------------------------------------------------
// small converts
// ---------------------------------------------------------------------------
__global__ __launch_bounds__(256) void f32_to_f16_kernel(const float* __restrict__ in,
                                                         _Float16* __restrict__ out, int n) {
  int i = blockIdx.x * 256 + threadIdx.x;
  if (i < n) out[i] = (_Float16)in[i];
}

__global__ __launch_bounds__(256) void neg_exp_kernel(const float* __restrict__ in,
                                                      float* __restrict__ out, int n) {
  int i = blockIdx.x * 256 + threadIdx.x;
  if (i < n) out[i] = -__expf(in[i]);
}

// ---------------------------------------------------------------------------
// JEGO gather: (b,k,l,c) -> element of desc0/desc1
// ---------------------------------------------------------------------------
__device__ __forceinline__ float jego_gather(const float* __restrict__ d0,
                                             const float* __restrict__ d1,
                                             int b, int k, int l, int c) {
  int h, w;
  const float* src;
  if (k == 0) {
    int r = l / 96, q = l - r * 96;
    h = 2 * r;
    if (q < 48) { w = 2 * q;      src = d0; }
    else        { w = 2 * q - 96; src = d1; }
  } else if (k == 1) {
    int r = l / 96, q = l - r * 96;
    w = 2 * r + 1;
    int hp = 2 * q + 1;
    if (hp < 96) { h = hp;      src = d0; }
    else         { h = hp - 96; src = d1; }
  } else if (k == 2) {
    int lp = (LL - 1) - l;
    int r = lp / 96, q = lp - r * 96;
    h = 2 * r;
    int w2 = 2 * q + 1;
    if (w2 < 96) { w = w2;      src = d0; }
    else         { w = w2 - 96; src = d1; }
  } else {
    int lp = (LL - 1) - l;
    int r = lp / 96, q = lp - r * 96;
    w = 2 * r;
    int hp = 2 * q + 1;
    if (hp < 96) { h = hp;      src = d0; }
    else         { h = hp - 96; src = d1; }
  }
  return src[(((size_t)b * CC + c) * HH + h) * WW + w];
}

// fused JEGO gather + LayerNorm(C=256) -> x_ln f16 [NTOK, 256]
__global__ __launch_bounds__(256) void gather_ln_kernel(const float* __restrict__ d0,
                                                        const float* __restrict__ d1,
                                                        const float* __restrict__ nw,
                                                        const float* __restrict__ nb,
                                                        _Float16* __restrict__ xln) {
  __shared__ float s[256];
  int tok = blockIdx.x;
  int n = tok / LL, l = tok - n * LL;
  int b = n >> 2, k = n & 3;
  int c = threadIdx.x;

  float v = jego_gather(d0, d1, b, k, l, c);

  s[c] = v;
  __syncthreads();
  #pragma unroll
  for (int off = 128; off > 0; off >>= 1) {
    if (c < off) s[c] += s[c + off];
    __syncthreads();
  }
  float mean = s[0] * (1.0f / 256.0f);
  __syncthreads();
  float dv = v - mean;
  s[c] = dv * dv;
  __syncthreads();
  #pragma unroll
  for (int off = 128; off > 0; off >>= 1) {
    if (c < off) s[c] += s[c + off];
    __syncthreads();
  }
  float var = s[0] * (1.0f / 256.0f);
  float y = dv * rsqrtf(var + 1e-5f) * nw[c] + nb[c];
  xln[(size_t)tok * CC + c] = (_Float16)y;
}

// ---------------------------------------------------------------------------
// WMMA GEMM:  C[M,N] (f32 acc) = A[M,K]f16 (row-major) * B[N,K]f16 (row-major)^T
// Block = 8 waves; each wave: one 16-row M tile x NTILES 16-col N tiles
// (NTILES compile-time -> no guards in the K loop, WMMAs issue back-to-back).
// K stepped by 32 with v_wmma_f32_16x16x32_f16.
// A sched_barrier between the fragment loads and the MAC burst keeps all
// loads of a k-step issued into distinct registers, so the hardware's
// in-order loadcnt produces partial waits and load/WMMA overlap.
// Fragment layouts used here (same contiguous k order for A and B):
//   A: lane%16 = row M; lane/16 selects K-half; elems 0..7 = K[kh*16..+7],
//      elems 8..15 = K[kh*16+8..+7]  (two contiguous 16B loads)
//   B: lane%16 = col N; elems 0..15 = K[(lane/16)*16 .. +15] of that column,
//      contiguous since B is stored [N,K] row-major (weights are (E,D))
//   D: col = lane%16, VGPR r -> row (lane/16)*8 + r
// ---------------------------------------------------------------------------
template <int NTILES, typename CT>
__global__ __launch_bounds__(256) void gemm_nt_wmma(const _Float16* __restrict__ A,
                                                    const _Float16* __restrict__ B,
                                                    CT* __restrict__ Cm,
                                                    int M, int N, int K) {
  const int lane  = threadIdx.x & 31;
  const int wave  = threadIdx.x >> 5;
  const int m0    = (blockIdx.x * 8 + wave) * 16;
  if (m0 >= M) return;                       // wave-uniform (grids are exact)
  const int n0b   = blockIdx.y * (16 * NTILES);
  const int kh    = lane >> 4;               // 0/1: K-half select
  const int i16   = lane & 15;               // A row / B,D col

  v8f acc[NTILES] = {};
  const _Float16* Arow = A + (size_t)(m0 + i16) * K;
  const _Float16* Brow[NTILES];
  #pragma unroll
  for (int t = 0; t < NTILES; ++t)
    Brow[t] = B + (size_t)(n0b + 16 * t + i16) * K + kh * 16;

  union F { v16h v; v8h h[2]; };

  #pragma unroll 2
  for (int k0 = 0; k0 < K; k0 += 32) {
    __builtin_prefetch(Arow + k0 + 64, 0, 1);   // global_prefetch_b8
    F af;
    af.h[0] = *(const v8h*)(Arow + k0 + kh * 16);
    af.h[1] = *(const v8h*)(Arow + k0 + kh * 16 + 8);
    F bf[NTILES];
    #pragma unroll
    for (int t = 0; t < NTILES; ++t) {
      bf[t].h[0] = *(const v8h*)(Brow[t] + k0);
      bf[t].h[1] = *(const v8h*)(Brow[t] + k0 + 8);
    }
    // Keep all fragment loads above, all WMMAs below: enables partial
    // s_wait_loadcnt instead of per-tile full waits.
    __builtin_amdgcn_sched_barrier(0);
    #pragma unroll
    for (int t = 0; t < NTILES; ++t) {
      acc[t] = __builtin_amdgcn_wmma_f32_16x16x32_f16(
          /*neg_a=*/false, af.v, /*neg_b=*/false, bf[t].v,
          /*c_mod=*/(short)0, acc[t], /*reuse_a=*/false, /*reuse_b=*/false);
    }
  }

  #pragma unroll
  for (int t = 0; t < NTILES; ++t) {
    const int n0 = n0b + 16 * t;
    #pragma unroll
    for (int r = 0; r < 8; ++r) {
      int row = m0 + kh * 8 + r;
      Cm[(size_t)row * N + n0 + i16] = (CT)acc[t][r];
    }
  }
}

// ---------------------------------------------------------------------------
// depthwise causal conv (D_CONV=4) + SiLU on xi = xz[:, :512] -> xc f16
// ---------------------------------------------------------------------------
__global__ __launch_bounds__(256) void conv_silu_kernel(const _Float16* __restrict__ xz,
                                                        const float* __restrict__ cw,
                                                        const float* __restrict__ cb,
                                                        _Float16* __restrict__ xc) {
  size_t idx = (size_t)blockIdx.x * 256 + threadIdx.x;
  const size_t total = (size_t)NTOK * DIN;
  if (idx >= total) return;
  int e = (int)(idx % DIN);
  size_t nl = idx / DIN;
  int l = (int)(nl % LL);
  size_t n = nl / LL;
  float acc = cb[e];
  #pragma unroll
  for (int t = 0; t < 4; ++t) {
    int ll = l - 3 + t;
    if (ll >= 0)
      acc += (float)xz[((size_t)n * LL + ll) * (2 * DIN) + e] * cw[e * 4 + t];
  }
  float sx = acc / (1.0f + __expf(-acc));    // silu
  xc[idx] = (_Float16)sx;
}

// ---------------------------------------------------------------------------
// selective scan, fused: dt = softplus(dbc[:, :16] @ Wdt^T + b),
// h[s] = h[s]*exp(dt*A[s]) + dt*x*B[s];  y = sum h*C + x*D;  y *= silu(z)
// one thread = one (n, e) channel, h[16] in registers, 4608 sequential steps.
// ---------------------------------------------------------------------------
__global__ __launch_bounds__(256) void scan_kernel(const float* __restrict__ dbc,
                                                   const _Float16* __restrict__ xc,
                                                   const _Float16* __restrict__ xz,
                                                   const float* __restrict__ Wdt,
                                                   const float* __restrict__ dtb,
                                                   const float* __restrict__ Aneg,
                                                   const float* __restrict__ Dp,
                                                   _Float16* __restrict__ yg) {
  int n = blockIdx.x;
  int e = blockIdx.y * 256 + threadIdx.x;

  float Areg[DST], Wreg[DTR], h[DST];
  #pragma unroll
  for (int s = 0; s < DST; ++s) { Areg[s] = Aneg[e * DST + s]; h[s] = 0.0f; }
  #pragma unroll
  for (int r = 0; r < DTR; ++r) Wreg[r] = Wdt[e * DTR + r];
  float bias = dtb[e], De = Dp[e];

  const float*    dp  = dbc + (size_t)n * LL * (DTR + 2 * DST);
  const _Float16* xcp = xc  + (size_t)n * LL * DIN + e;
  const _Float16* zp  = xz  + (size_t)n * LL * (2 * DIN) + DIN + e;
  _Float16*       yp  = yg  + (size_t)n * LL * DIN + e;

  for (int l = 0; l < LL; ++l) {
    const float* row = dp + (size_t)l * (DTR + 2 * DST);
    float dl = bias;
    #pragma unroll
    for (int r = 0; r < DTR; ++r) dl += row[r] * Wreg[r];
    float dt = (dl > 20.0f) ? dl : log1pf(__expf(dl));   // softplus
    float xt = (float)xcp[(size_t)l * DIN];
    float dx = dt * xt;
    float y = 0.0f;
    #pragma unroll
    for (int s = 0; s < DST; ++s) {
      float dA = __expf(dt * Areg[s]);
      h[s] = h[s] * dA + dx * row[DTR + s];          // B_t
      y += h[s] * row[DTR + DST + s];                // C_t
    }
    y += xt * De;
    float z = (float)zp[(size_t)l * (2 * DIN)];
    float g = z / (1.0f + __expf(-z));               // silu(z)
    yp[(size_t)l * DIN] = (_Float16)(y * g);
  }
}

// ---------------------------------------------------------------------------
// inverse JEGO merge: out[d,b,c,h,w] = 2*desc_d[...] + mamba_out[b,k,l,c]
// (residual token == source desc element, so o = desc + mamba_out)
// ---------------------------------------------------------------------------
__global__ __launch_bounds__(256) void merge_kernel(const float* __restrict__ d0,
                                                    const float* __restrict__ d1,
                                                    const float* __restrict__ mout,
                                                    float* __restrict__ out) {
  size_t idx = (size_t)blockIdx.x * 256 + threadIdx.x;
  const size_t total = (size_t)2 * BB * CC * HH * WW;
  if (idx >= total) return;
  int w = (int)(idx % WW); size_t t = idx / WW;
  int h = (int)(t % HH);   t /= HH;
  int c = (int)(t % CC);   t /= CC;
  int b = (int)(t % BB);
  int d = (int)(t / BB);

  int k, l;
  bool he = (h & 1) == 0, we = (w & 1) == 0;
  if (he && we) {          // k=0
    k = 0;
    int r = h >> 1;
    int q = (d == 0) ? (w >> 1) : (48 + (w >> 1));
    l = r * 96 + q;
  } else if (he) {         // h even, w odd -> k=2 (reversed)
    k = 2;
    int r = h >> 1;
    int q = (d == 0) ? ((w - 1) >> 1) : ((w + 95) >> 1);
    l = (LL - 1) - (r * 96 + q);
  } else if (!we) {        // h odd, w odd -> k=1
    k = 1;
    int r = (w - 1) >> 1;
    int q = (d == 0) ? ((h - 1) >> 1) : ((h + 95) >> 1);
    l = r * 96 + q;
  } else {                 // h odd, w even -> k=3 (reversed)
    k = 3;
    int r = w >> 1;
    int q = (d == 0) ? ((h - 1) >> 1) : ((h + 95) >> 1);
    l = (LL - 1) - (r * 96 + q);
  }

  const float* desc = (d == 0) ? d0 : d1;
  float dv = desc[(((size_t)b * CC + c) * HH + h) * WW + w];
  int nn = b * KK + k;
  float mo = mout[((size_t)nn * LL + l) * CC + c];
  out[idx] = 2.0f * dv + mo;
}

// ---------------------------------------------------------------------------
// launch
// ---------------------------------------------------------------------------
extern "C" void kernel_launch(void* const* d_in, const int* in_sizes, int n_in,
                              void* d_out, int out_size, void* d_ws, size_t ws_size,
                              hipStream_t stream) {
  const float* desc0     = (const float*)d_in[0];
  const float* desc1     = (const float*)d_in[1];
  const float* norm_w    = (const float*)d_in[2];
  const float* norm_b    = (const float*)d_in[3];
  const float* in_proj_w = (const float*)d_in[4];   // (1024, 256)
  const float* conv_w    = (const float*)d_in[5];   // (512, 4)
  const float* conv_b    = (const float*)d_in[6];   // (512)
  const float* x_proj_w  = (const float*)d_in[7];   // (48, 512)
  const float* dt_proj_w = (const float*)d_in[8];   // (512, 16)
  const float* dt_proj_b = (const float*)d_in[9];   // (512)
  const float* A_log     = (const float*)d_in[10];  // (512, 16)
  const float* D_param   = (const float*)d_in[11];  // (512)
  const float* out_proj_w= (const float*)d_in[12];  // (256, 512)
  float* outp = (float*)d_out;

  char* ws = (char*)d_ws;
  // workspace layout (bytes); peak ~355 MB
  const size_t o_win  = 0;                                  // 1024*256 f16
  const size_t o_wxp  = o_win  + (size_t)1024 * 256 * 2;    // 48*512  f16
  const size_t o_wout = o_wxp  + (size_t)48 * 512 * 2;      // 256*512 f16
  const size_t o_aneg = o_wout + (size_t)256 * 512 * 2;     // 512*16  f32
  const size_t o_xln  = 1ull << 20;                         // NTOK*256 f16   (37.7 MB)
  const size_t o_xz   = o_xln + (size_t)NTOK * CC * 2;      // NTOK*1024 f16  (151 MB)
  const size_t o_xc   = o_xz  + (size_t)NTOK * 2 * DIN * 2; // NTOK*512 f16   (75.5 MB)
  const size_t o_dbc  = o_xc  + (size_t)NTOK * DIN * 2;     // NTOK*48 f32    (14.2 MB)
  const size_t o_yg   = o_dbc + (size_t)NTOK * 48 * 4;      // NTOK*512 f16   (75.5 MB)
  const size_t o_mout = o_xz;  // alias: xz dead after scan; NTOK*256 f32 fits

  _Float16* Win16  = (_Float16*)(ws + o_win);
  _Float16* Wxp16  = (_Float16*)(ws + o_wxp);
  _Float16* Wout16 = (_Float16*)(ws + o_wout);
  float*    Aneg   = (float*)   (ws + o_aneg);
  _Float16* xln    = (_Float16*)(ws + o_xln);
  _Float16* xz16   = (_Float16*)(ws + o_xz);
  _Float16* xc16   = (_Float16*)(ws + o_xc);
  float*    dbc    = (float*)   (ws + o_dbc);
  _Float16* yg16   = (_Float16*)(ws + o_yg);
  float*    mout   = (float*)   (ws + o_mout);

  // 1) weight preparation
  f32_to_f16_kernel<<<(1024 * 256 + 255) / 256, 256, 0, stream>>>(in_proj_w, Win16, 1024 * 256);
  f32_to_f16_kernel<<<(48 * 512 + 255) / 256, 256, 0, stream>>>(x_proj_w, Wxp16, 48 * 512);
  f32_to_f16_kernel<<<(256 * 512 + 255) / 256, 256, 0, stream>>>(out_proj_w, Wout16, 256 * 512);
  neg_exp_kernel<<<(512 * 16 + 255) / 256, 256, 0, stream>>>(A_log, Aneg, 512 * 16);

  // 2) JEGO gather + LayerNorm
  gather_ln_kernel<<<NTOK, 256, 0, stream>>>(desc0, desc1, norm_w, norm_b, xln);

  // 3) in_proj GEMM: [73728,256] x [1024,256]^T -> xz f16 [73728,1024]
  //    wave tile 16x128 (8 N-tiles), grid (576, 8)
  gemm_nt_wmma<8, _Float16><<<dim3(NTOK / 16 / 8, 1024 / 128), 256, 0, stream>>>(
      xln, Win16, xz16, NTOK, 1024, 256);

  // 4) depthwise conv + silu -> xc f16
  conv_silu_kernel<<<(unsigned)(((size_t)NTOK * DIN + 255) / 256), 256, 0, stream>>>(
      xz16, conv_w, conv_b, xc16);

  // 5) x_proj GEMM: [73728,512] x [48,512]^T -> dbc f32 [73728,48]  (3 N-tiles)
  gemm_nt_wmma<3, float><<<dim3(NTOK / 16 / 8, 1), 256, 0, stream>>>(
      xc16, Wxp16, dbc, NTOK, 48, 512);

  // 6) fused dt + selective scan + gating -> y_gated f16
  scan_kernel<<<dim3(NN, DIN / 256), 256, 0, stream>>>(
      dbc, xc16, xz16, dt_proj_w, dt_proj_b, Aneg, D_param, yg16);

  // 7) out_proj GEMM: [73728,512] x [256,512]^T -> mamba_out f32 [73728,256]
  //    wave tile 16x128 (8 N-tiles), grid (576, 2)
  gemm_nt_wmma<8, float><<<dim3(NTOK / 16 / 8, 256 / 128), 256, 0, stream>>>(
      yg16, Wout16, mout, NTOK, 256, 512);

  // 8) inverse JEGO merge + residual: out = 2*desc + mamba_out
  merge_kernel<<<(unsigned)(((size_t)2 * BB * CC * HH * WW + 255) / 256), 256, 0, stream>>>(
      desc0, desc1, mout, outp);
}
